// TransformerModel_18305150616352
// MI455X (gfx1250) — matrix operand order repaired and checked by reading, not compiled, not run on hardware
//
#include <hip/hip_runtime.h>

// Model dims (fixed by reference)
#define E_DIM 1024
#define NHEAD 16
#define HDIM  64
#define HID_  4096
#define L_    6
#define V_    32000
#define B_    4
#define S_    512
#define T_    512

typedef __attribute__((ext_vector_type(16))) _Float16 v16h;
typedef __attribute__((ext_vector_type(8)))  _Float16 v8h;
typedef __attribute__((ext_vector_type(8)))  float    v8f;
typedef __attribute__((ext_vector_type(4)))  int      v4i;

enum { GF_BIAS = 1, GF_RELU = 2, GF_RES = 4, GF_OUT16 = 8, GF_VT16 = 16, GF_SHADOW = 32 };

#if defined(__has_builtin)
#if __has_builtin(__builtin_amdgcn_global_load_async_to_lds_b128) && \
    __has_builtin(__builtin_amdgcn_s_wait_asynccnt)
#define HAVE_ASYNC_LDS 1
#endif
#endif

#if defined(HAVE_ASYNC_LDS)
// signature (probe-derived): (v4i* global_src, as3 v4i* lds_dst, int offset, int cpol)
__device__ __forceinline__ void async_cp16(const _Float16* g, _Float16* l)
{
    __builtin_amdgcn_global_load_async_to_lds_b128(
        (v4i*)g, (__attribute__((address_space(3))) v4i*)l, 0, 0);
}
#endif

// ---------------------------------------------------------------------------
// Weight convert+transpose: in f32 [K,N] row-major -> out f16 [N,K] row-major.
// One matrix per blockIdx.z (stride K*N). 32x32 tiles via LDS.
// ---------------------------------------------------------------------------
__global__ __launch_bounds__(256)
void transpose_cvt_kernel(const float* __restrict__ in, _Float16* __restrict__ out,
                          int K, int N)
{
    __shared__ float tile[32][33];
    const long mat = blockIdx.z;
    in  += mat * (long)K * N;
    out += mat * (long)K * N;
    const int n0 = blockIdx.x * 32;
    const int k0 = blockIdx.y * 32;
    const int tx = threadIdx.x & 31;
    const int ty = threadIdx.x >> 5;          // 0..7
#pragma unroll
    for (int j = 0; j < 4; ++j)
        tile[ty + j * 8][tx] = in[(long)(k0 + ty + j * 8) * N + n0 + tx];
    __syncthreads();
#pragma unroll
    for (int j = 0; j < 4; ++j)
        out[(long)(n0 + ty + j * 8) * K + k0 + tx] = (_Float16)tile[tx][ty + j * 8];
}

// ---------------------------------------------------------------------------
// Embedding + sinusoidal positional encoding (f32 out)
// ---------------------------------------------------------------------------
__global__ __launch_bounds__(256)
void embed_pe_kernel(const int* __restrict__ toks, const float* __restrict__ emb,
                     float* __restrict__ out, int slen, float scale)
{
    const int token = blockIdx.x;
    const int pos   = token % slen;
    const int id    = toks[token];
    const float kdiv = -logf(10000.f) / (float)E_DIM;
    for (int c = threadIdx.x; c < E_DIM; c += 256) {
        int   i2  = c & ~1;
        float ang = (float)pos * expf((float)i2 * kdiv);
        float pe  = (c & 1) ? cosf(ang) : sinf(ang);
        out[(long)token * E_DIM + c] = emb[(long)id * E_DIM + c] * scale + pe;
    }
}

// ---------------------------------------------------------------------------
// LayerNorm: f32 in -> f32 out + f16 shadow (GEMM operand)
// ---------------------------------------------------------------------------
__global__ __launch_bounds__(256)
void layernorm_kernel(const float* __restrict__ X, const float* __restrict__ g,
                      const float* __restrict__ b, float* __restrict__ O,
                      _Float16* __restrict__ O16)
{
    __shared__ float red[256];
    const long row = blockIdx.x;
    const float* x = X + row * E_DIM;
    float v[4];
    float s = 0.f;
#pragma unroll
    for (int i = 0; i < 4; ++i) { v[i] = x[threadIdx.x + i * 256]; s += v[i]; }
    red[threadIdx.x] = s; __syncthreads();
    for (int off = 128; off > 0; off >>= 1) {
        if (threadIdx.x < off) red[threadIdx.x] += red[threadIdx.x + off];
        __syncthreads();
    }
    const float mu = red[0] * (1.f / E_DIM);
    __syncthreads();
    float ss = 0.f;
#pragma unroll
    for (int i = 0; i < 4; ++i) { float d = v[i] - mu; ss += d * d; }
    red[threadIdx.x] = ss; __syncthreads();
    for (int off = 128; off > 0; off >>= 1) {
        if (threadIdx.x < off) red[threadIdx.x] += red[threadIdx.x + off];
        __syncthreads();
    }
    const float rstd = rsqrtf(red[0] * (1.f / E_DIM) + 1e-5f);
    float* o = O + row * E_DIM;
    _Float16* o16 = O16 + row * E_DIM;
#pragma unroll
    for (int i = 0; i < 4; ++i) {
        int c = threadIdx.x + i * 256;
        float val = (v[i] - mu) * rstd * g[c] + b[c];
        o[c] = val;
        o16[c] = (_Float16)val;
    }
}

// ---------------------------------------------------------------------------
// Masked softmax: reads f32 scores, writes f16 probabilities (GEMM operand)
// ---------------------------------------------------------------------------
__global__ __launch_bounds__(256)
void softmax_mask_kernel(const float* __restrict__ Sc, _Float16* __restrict__ P16,
                         const int* __restrict__ qtok, const int* __restrict__ ktok,
                         int mode, int Tq, int Tk)
{
    __shared__ float red[256];
    const int q = blockIdx.x;
    const int z = blockIdx.y;            // b*NHEAD + h
    const int b = z / NHEAD;
    const float* row = Sc + ((long)z * Tq + q) * Tk;
    _Float16* prow = P16 + ((long)z * Tq + q) * Tk;
    const bool qvalid = qtok[b * Tq + q] != 0;
    float vals[2];
    float mx = -3.0e38f;
#pragma unroll
    for (int i = 0; i < 2; ++i) {
        int k = threadIdx.x + i * 256;
        bool valid = qvalid;
        if (mode == 0)      valid = valid && (ktok[b * Tk + k] != 0);
        else if (mode == 1) valid = valid && (k <= q);
        float s = valid ? row[k] : -10000.f;
        vals[i] = s;
        mx = fmaxf(mx, s);
    }
    red[threadIdx.x] = mx; __syncthreads();
    for (int off = 128; off > 0; off >>= 1) {
        if (threadIdx.x < off) red[threadIdx.x] = fmaxf(red[threadIdx.x], red[threadIdx.x + off]);
        __syncthreads();
    }
    mx = red[0]; __syncthreads();
    float sum = 0.f;
#pragma unroll
    for (int i = 0; i < 2; ++i) { vals[i] = expf(vals[i] - mx); sum += vals[i]; }
    red[threadIdx.x] = sum; __syncthreads();
    for (int off = 128; off > 0; off >>= 1) {
        if (threadIdx.x < off) red[threadIdx.x] += red[threadIdx.x + off];
        __syncthreads();
    }
    const float inv = 1.f / red[0];
#pragma unroll
    for (int i = 0; i < 2; ++i) prow[threadIdx.x + i * 256] = (_Float16)(vals[i] * inv);
}

// ---------------------------------------------------------------------------
// Batched WMMA GEMM, all-f16 operands:
//   C[z] = alpha * A[z] (MxK, row-major) x B[z]^T (B stored [N,K] row-major)
// Tile 128x64x32, 256 threads = 8 wave32s, wave w owns one 16-row M-subtile
// and 4 accumulators. Tiles are copied to LDS verbatim (raw f16 rows), so
// fragments are contiguous LDS vector loads:
//   A frag: two 16B runs  (k = fhi*8..+7 and 16+fhi*8..+7), row = 16w+fm
//   B frag: two 16B runs  (k = fhi*8..+7 and 16+fhi*8..+7),                row = 16sub+fm
// Staging uses GLOBAL_LOAD_ASYNC_TO_LDS_B128 (ASYNCcnt) when the builtin is
// available, else register float4 copies. Double-buffered LDS.
// ---------------------------------------------------------------------------
#define BM 128
#define BN 64
#define BK 32
#define A_HALVES (BM * BK)
#define B_HALVES (BN * BK)

__global__ __launch_bounds__(256)
void wmma_gemm_kernel(const _Float16* __restrict__ A, const _Float16* __restrict__ Bm,
                      float* __restrict__ C, _Float16* __restrict__ C16,
                      const float* __restrict__ bias, const float* __restrict__ Res,
                      int K, int lda, int ldb, int ldc,
                      int zdiv,
                      long sAh, long sAb, long sBh, long sBb,
                      long sCh, long sCb, long sRh, long sRb,
                      float alpha, int flags)
{
    const int z  = blockIdx.z;
    const int zh = z % zdiv;
    const int zb = z / zdiv;
    A  += zh * sAh + zb * sAb;
    Bm += zh * sBh + zb * sBb;
    const long coff = zh * sCh + zb * sCb;
    if (C)   C   += coff;
    if (C16) C16 += coff;
    const float* R = (flags & GF_RES) ? (Res + zh * sRh + zb * sRb) : nullptr;

    const int tile_n0 = blockIdx.x * BN;
    const int tile_m0 = blockIdx.y * BM;

    __shared__ __align__(16) _Float16 Af[2][A_HALVES];   // 128 rows x 32 k
    __shared__ __align__(16) _Float16 Bf[2][B_HALVES];   // 64 rows (n) x 32 k

    const int tid  = threadIdx.x;
    const int wave = tid >> 5;
    const int lane = tid & 31;
    const int fm   = lane & 15;
    const int fhi  = lane >> 4;

    v8f acc[4];
#pragma unroll
    for (int i = 0; i < 4; ++i) acc[i] = (v8f){};

    const int nk = K / BK;

#if defined(HAVE_ASYNC_LDS)
    // --- async-to-LDS staging path (ASYNCcnt) ---
    {
        // A: 512 16B chunks -> 2/thread ; B: 256 chunks -> 1/thread
#pragma unroll
        for (int e = 0; e < 2; ++e) {
            int idx = tid + e * 256;
            int r = idx >> 2, c = (idx & 3) * 8;
            async_cp16(&A[(long)(tile_m0 + r) * lda + c], &Af[0][r * 32 + c]);
        }
        {
            int r = tid >> 2, c = (tid & 3) * 8;
            async_cp16(&Bm[(long)(tile_n0 + r) * ldb + c], &Bf[0][r * 32 + c]);
        }
    }
    for (int kt = 0; kt < nk; ++kt) {
        const int buf = kt & 1;
        __builtin_amdgcn_s_wait_asynccnt(0);
        __syncthreads();
        if (kt + 1 < nk) {
            const int k0 = (kt + 1) * BK;
#pragma unroll
            for (int e = 0; e < 2; ++e) {
                int idx = tid + e * 256;
                int r = idx >> 2, c = (idx & 3) * 8;
                async_cp16(&A[(long)(tile_m0 + r) * lda + k0 + c], &Af[buf ^ 1][r * 32 + c]);
            }
            {
                int r = tid >> 2, c = (tid & 3) * 8;
                async_cp16(&Bm[(long)(tile_n0 + r) * ldb + k0 + c], &Bf[buf ^ 1][r * 32 + c]);
            }
        }
        const int arow = wave * 16 + fm;
        v8h alo = *(const v8h*)&Af[buf][arow * 32 + fhi * 8];
        v8h ahi = *(const v8h*)&Af[buf][arow * 32 + 16 + fhi * 8];
        v16h afrag = __builtin_shufflevector(alo, ahi, 0,1,2,3,4,5,6,7,8,9,10,11,12,13,14,15);
        v16h bfr[4];
#pragma unroll
        for (int sub = 0; sub < 4; ++sub) {
            int brow = sub * 16 + fm;
            v8h blo = *(const v8h*)&Bf[buf][brow * 32 + fhi * 8];
            v8h bhi = *(const v8h*)&Bf[buf][brow * 32 + 16 + fhi * 8];
            bfr[sub] = __builtin_shufflevector(blo, bhi, 0,1,2,3,4,5,6,7,8,9,10,11,12,13,14,15);
        }
#pragma unroll
        for (int sub = 0; sub < 4; ++sub)
            acc[sub] = __builtin_amdgcn_wmma_f32_16x16x32_f16(false, afrag, false, bfr[sub],
                                                              (short)0, acc[sub], false, false);
        __syncthreads();
    }
#else
    // --- register-staged fallback (float4 copies, pipelined) ---
    uint4 ra[2], rb1;
    {
#pragma unroll
        for (int e = 0; e < 2; ++e) {
            int idx = tid + e * 256;
            int r = idx >> 2, c = (idx & 3) * 8;
            ra[e] = *(const uint4*)&A[(long)(tile_m0 + r) * lda + c];
        }
        int r = tid >> 2, c = (tid & 3) * 8;
        rb1 = *(const uint4*)&Bm[(long)(tile_n0 + r) * ldb + c];
    }
    for (int kt = 0; kt < nk; ++kt) {
        const int buf = kt & 1;
#pragma unroll
        for (int e = 0; e < 2; ++e) {
            int idx = tid + e * 256;
            int r = idx >> 2, c = (idx & 3) * 8;
            *(uint4*)&Af[buf][r * 32 + c] = ra[e];
        }
        {
            int r = tid >> 2, c = (tid & 3) * 8;
            *(uint4*)&Bf[buf][r * 32 + c] = rb1;
        }
        __syncthreads();
        if (kt + 1 < nk) {
            const int k0 = (kt + 1) * BK;
#pragma unroll
            for (int e = 0; e < 2; ++e) {
                int idx = tid + e * 256;
                int r = idx >> 2, c = (idx & 3) * 8;
                ra[e] = *(const uint4*)&A[(long)(tile_m0 + r) * lda + k0 + c];
            }
            int r = tid >> 2, c = (tid & 3) * 8;
            rb1 = *(const uint4*)&Bm[(long)(tile_n0 + r) * ldb + k0 + c];
        }
        const int arow = wave * 16 + fm;
        v8h alo = *(const v8h*)&Af[buf][arow * 32 + fhi * 8];
        v8h ahi = *(const v8h*)&Af[buf][arow * 32 + 16 + fhi * 8];
        v16h afrag = __builtin_shufflevector(alo, ahi, 0,1,2,3,4,5,6,7,8,9,10,11,12,13,14,15);
        v16h bfr[4];
#pragma unroll
        for (int sub = 0; sub < 4; ++sub) {
            int brow = sub * 16 + fm;
            v8h blo = *(const v8h*)&Bf[buf][brow * 32 + fhi * 8];
            v8h bhi = *(const v8h*)&Bf[buf][brow * 32 + 16 + fhi * 8];
            bfr[sub] = __builtin_shufflevector(blo, bhi, 0,1,2,3,4,5,6,7,8,9,10,11,12,13,14,15);
        }
#pragma unroll
        for (int sub = 0; sub < 4; ++sub)
            acc[sub] = __builtin_amdgcn_wmma_f32_16x16x32_f16(false, afrag, false, bfr[sub],
                                                              (short)0, acc[sub], false, false);
    }
#endif

    // Epilogue. C/D layout: lane<16 -> M=r, N=lane ; lane>=16 -> M=r+8, N=lane-16
#pragma unroll
    for (int sub = 0; sub < 4; ++sub) {
#pragma unroll
        for (int r = 0; r < 8; ++r) {
            int row = tile_m0 + wave * 16 + r + fhi * 8;
            int col = tile_n0 + sub * 16 + fm;
            float v = acc[sub][r] * alpha;
            if (flags & GF_BIAS) v += bias[col];
            if (flags & GF_RELU) v = fmaxf(v, 0.f);
            if (flags & GF_RES)  v += R[(long)row * ldc + col];
            if (flags & GF_VT16) {
                // V-projection transpose: [b,h,d,s] so ctx GEMM consumes [N,K]
                long idx = ((long)(row >> 9) * NHEAD + (col >> 6)) * (HDIM * S_)
                         + (long)(col & (HDIM - 1)) * S_ + (row & (S_ - 1));
                C16[idx] = (_Float16)v;
            } else if (flags & GF_OUT16) {
                C16[(long)row * ldc + col] = (_Float16)v;
            } else {
                C[(long)row * ldc + col] = v;
                if (flags & GF_SHADOW) C16[(long)row * ldc + col] = (_Float16)v;
            }
        }
    }
}

// ---------------------------------------------------------------------------
// Host-side launch helper
// ---------------------------------------------------------------------------
static void launch_gemm(hipStream_t st, const _Float16* A, const _Float16* Bm,
                        float* C, _Float16* C16,
                        const float* bias, const float* Res,
                        int M, int N, int K, int lda, int ldb, int ldc,
                        float alpha, int flags,
                        int batches = 1, int zdiv = 1,
                        long sAh = 0, long sAb = 0, long sBh = 0, long sBb = 0,
                        long sCh = 0, long sCb = 0, long sRh = 0, long sRb = 0)
{
    dim3 grid((N + BN - 1) / BN, (M + BM - 1) / BM, batches);
    wmma_gemm_kernel<<<grid, 256, 0, st>>>(A, Bm, C, C16, bias, Res,
                                           K, lda, ldb, ldc,
                                           zdiv, sAh, sAb, sBh, sBb,
                                           sCh, sCb, sRh, sRb, alpha, flags);
}

extern "C" void kernel_launch(void* const* d_in, const int* in_sizes, int n_in,
                              void* d_out, int out_size, void* d_ws, size_t ws_size,
                              hipStream_t stream)
{
    (void)in_sizes; (void)n_in; (void)out_size; (void)ws_size;

    const int*   src          = (const int*)  d_in[0];
    const int*   tgt          = (const int*)  d_in[1];
    const float* emb_enc      = (const float*)d_in[2];
    const float* emb_dec      = (const float*)d_in[3];
    const float* enc_qkv      = (const float*)d_in[4];
    const float* enc_w1       = (const float*)d_in[5];
    const float* enc_b1       = (const float*)d_in[6];
    const float* enc_w2       = (const float*)d_in[7];
    const float* enc_b2       = (const float*)d_in[8];
    const float* enc_ln_g     = (const float*)d_in[9];
    const float* enc_ln_b     = (const float*)d_in[10];
    const float* dec_qkv_self = (const float*)d_in[11];
    const float* dec_qkv_cross= (const float*)d_in[12];
    const float* dec_w1       = (const float*)d_in[13];
    const float* dec_b1       = (const float*)d_in[14];
    const float* dec_w2       = (const float*)d_in[15];
    const float* dec_b2       = (const float*)d_in[16];
    const float* dec_ln_g     = (const float*)d_in[17];
    const float* dec_ln_b     = (const float*)d_in[18];
    const float* fin_ln_g     = (const float*)d_in[19];
    const float* fin_ln_b     = (const float*)d_in[20];
    const float* out_w        = (const float*)d_in[21];
    const float* out_b        = (const float*)d_in[22];

    const int M = B_ * S_;                  // 2048 rows (B*S == B*T)
    const long nTok = (long)M * E_DIM;      // 2M elems

    // ---- workspace carve: f32 region then f16 region ----
    float* f = (float*)d_ws;
    float* X    = f; f += nTok;                         // encoder state
    float* Y    = f; f += nTok;                         // decoder state
    float* XN   = f; f += nTok;                         // LN out (residual source)
    float* Abuf = f; f += nTok;                         // attn out (pre-LN)
    float* SC   = f; f += (long)B_ * NHEAD * S_ * S_;   // raw attention scores

    _Float16* h = (_Float16*)f;
    _Float16* XN16  = h; h += nTok;                     // LN f16 shadow
    _Float16* X16   = h; h += nTok;                     // encoder-out f16 shadow
    _Float16* Q16   = h; h += nTok;
    _Float16* K16   = h; h += nTok;
    _Float16* VT16  = h; h += nTok;                     // [b,h,d,s]
    _Float16* P16   = h; h += (long)B_ * NHEAD * S_ * S_;
    _Float16* H116  = h; h += (long)M * HID_;
    _Float16* WqkvE = h; h += (long)3 * L_ * E_DIM * E_DIM;
    _Float16* WqkvS = h; h += (long)3 * L_ * E_DIM * E_DIM;
    _Float16* WqkvC = h; h += (long)3 * L_ * E_DIM * E_DIM;
    _Float16* W1e   = h; h += (long)L_ * E_DIM * HID_;
    _Float16* W2e   = h; h += (long)L_ * HID_ * E_DIM;
    _Float16* W1d   = h; h += (long)L_ * E_DIM * HID_;
    _Float16* W2d   = h; h += (long)L_ * HID_ * E_DIM;
    _Float16* WoutT = h; h += (long)E_DIM * V_;

    // ---- one-time weight convert + transpose to f16 [N,K] ----
    transpose_cvt_kernel<<<dim3(E_DIM/32, E_DIM/32, 3*L_), 256, 0, stream>>>(enc_qkv,       WqkvE, E_DIM, E_DIM);
    transpose_cvt_kernel<<<dim3(E_DIM/32, E_DIM/32, 3*L_), 256, 0, stream>>>(dec_qkv_self,  WqkvS, E_DIM, E_DIM);
    transpose_cvt_kernel<<<dim3(E_DIM/32, E_DIM/32, 3*L_), 256, 0, stream>>>(dec_qkv_cross, WqkvC, E_DIM, E_DIM);
    transpose_cvt_kernel<<<dim3(HID_/32,  E_DIM/32, L_),   256, 0, stream>>>(enc_w1, W1e, E_DIM, HID_);
    transpose_cvt_kernel<<<dim3(E_DIM/32, HID_/32,  L_),   256, 0, stream>>>(enc_w2, W2e, HID_, E_DIM);
    transpose_cvt_kernel<<<dim3(HID_/32,  E_DIM/32, L_),   256, 0, stream>>>(dec_w1, W1d, E_DIM, HID_);
    transpose_cvt_kernel<<<dim3(E_DIM/32, HID_/32,  L_),   256, 0, stream>>>(dec_w2, W2d, HID_, E_DIM);
    transpose_cvt_kernel<<<dim3(V_/32,    E_DIM/32, 1),    256, 0, stream>>>(out_w, WoutT, E_DIM, V_);

    const long sE   = (long)S_ * E_DIM;        // per-batch stride in token bufs
    const long sSch = (long)S_ * S_;           // per-head stride in scores
    const long sScb = (long)NHEAD * S_ * S_;   // per-batch stride in scores
    const long sVh  = (long)HDIM * S_;         // per-head stride in VT16
    const long sVb  = (long)NHEAD * HDIM * S_;
    const float inv_sqrt_e = 0.03125f;         // 1/sqrt(1024)

    // ---- embeddings ----
    embed_pe_kernel<<<M, 256, 0, stream>>>(src, emb_enc, X, S_, 32.0f); // sqrt(E)
    embed_pe_kernel<<<M, 256, 0, stream>>>(tgt, emb_dec, Y, T_, 1.0f);

    // ---- encoder ----
    for (int i = 0; i < L_; ++i) {
        const float* g  = enc_ln_g + (long)i * E_DIM;
        const float* bb = enc_ln_b + (long)i * E_DIM;
        layernorm_kernel<<<M, 256, 0, stream>>>(X, g, bb, XN, XN16);
        const _Float16* wq = WqkvE + ((long)i * 3 + 0) * E_DIM * E_DIM;
        const _Float16* wk = WqkvE + ((long)i * 3 + 1) * E_DIM * E_DIM;
        const _Float16* wv = WqkvE + ((long)i * 3 + 2) * E_DIM * E_DIM;
        launch_gemm(stream, XN16, wq, nullptr, Q16, nullptr, nullptr, M, E_DIM, E_DIM, E_DIM, E_DIM, E_DIM, 1.f, GF_OUT16);
        launch_gemm(stream, XN16, wk, nullptr, K16, nullptr, nullptr, M, E_DIM, E_DIM, E_DIM, E_DIM, E_DIM, 1.f, GF_OUT16);
        launch_gemm(stream, XN16, wv, nullptr, VT16, nullptr, nullptr, M, E_DIM, E_DIM, E_DIM, E_DIM, E_DIM, 1.f, GF_VT16);
        // scores = Q x K^T / sqrt(E)   per (b,h)
        launch_gemm(stream, Q16, K16, SC, nullptr, nullptr, nullptr, S_, S_, HDIM, E_DIM, E_DIM, S_,
                    inv_sqrt_e, 0, B_ * NHEAD, NHEAD,
                    HDIM, sE, HDIM, sE, sSch, sScb, 0, 0);
        softmax_mask_kernel<<<dim3(S_, B_ * NHEAD), 256, 0, stream>>>(SC, P16, src, src, 0, S_, S_);
        // ctx = P x V^T(stored [d,s]), fused residual (+XN)
        launch_gemm(stream, P16, VT16, Abuf, nullptr, nullptr, XN, S_, HDIM, S_, S_, S_, E_DIM,
                    1.f, GF_RES, B_ * NHEAD, NHEAD,
                    sSch, sScb, sVh, sVb, HDIM, sE, HDIM, sE);
        layernorm_kernel<<<M, 256, 0, stream>>>(Abuf, g, bb, XN, XN16);
        launch_gemm(stream, XN16, W1e + (long)i * E_DIM * HID_, nullptr, H116,
                    enc_b1 + (long)i * HID_, nullptr,
                    M, HID_, E_DIM, E_DIM, E_DIM, HID_, 1.f, GF_BIAS | GF_RELU | GF_OUT16);
        launch_gemm(stream, H116, W2e + (long)i * HID_ * E_DIM, X, X16,
                    enc_b2 + (long)i * E_DIM, XN,
                    M, E_DIM, HID_, HID_, HID_, E_DIM, 1.f, GF_BIAS | GF_RES | GF_SHADOW);
    }

    // ---- decoder ----
    for (int i = 0; i < L_; ++i) {
        const float* g  = dec_ln_g + (long)i * E_DIM;
        const float* bb = dec_ln_b + (long)i * E_DIM;
        // masked self attention
        layernorm_kernel<<<M, 256, 0, stream>>>(Y, g, bb, XN, XN16);
        const _Float16* wq = WqkvS + ((long)i * 3 + 0) * E_DIM * E_DIM;
        const _Float16* wk = WqkvS + ((long)i * 3 + 1) * E_DIM * E_DIM;
        const _Float16* wv = WqkvS + ((long)i * 3 + 2) * E_DIM * E_DIM;
        launch_gemm(stream, XN16, wq, nullptr, Q16, nullptr, nullptr, M, E_DIM, E_DIM, E_DIM, E_DIM, E_DIM, 1.f, GF_OUT16);
        launch_gemm(stream, XN16, wk, nullptr, K16, nullptr, nullptr, M, E_DIM, E_DIM, E_DIM, E_DIM, E_DIM, 1.f, GF_OUT16);
        launch_gemm(stream, XN16, wv, nullptr, VT16, nullptr, nullptr, M, E_DIM, E_DIM, E_DIM, E_DIM, E_DIM, 1.f, GF_VT16);
        launch_gemm(stream, Q16, K16, SC, nullptr, nullptr, nullptr, T_, T_, HDIM, E_DIM, E_DIM, T_,
                    inv_sqrt_e, 0, B_ * NHEAD, NHEAD,
                    HDIM, sE, HDIM, sE, sSch, sScb, 0, 0);
        softmax_mask_kernel<<<dim3(T_, B_ * NHEAD), 256, 0, stream>>>(SC, P16, tgt, tgt, 1, T_, T_);
        launch_gemm(stream, P16, VT16, Abuf, nullptr, nullptr, XN, T_, HDIM, T_, T_, S_, E_DIM,
                    1.f, GF_RES, B_ * NHEAD, NHEAD,
                    sSch, sScb, sVh, sVb, HDIM, sE, HDIM, sE);
        layernorm_kernel<<<M, 256, 0, stream>>>(Abuf, g, bb, XN, XN16);
        // cross attention: queries from XN16, keys/values from encoder output X16
        const _Float16* cwq = WqkvC + ((long)i * 3 + 0) * E_DIM * E_DIM;
        const _Float16* cwk = WqkvC + ((long)i * 3 + 1) * E_DIM * E_DIM;
        const _Float16* cwv = WqkvC + ((long)i * 3 + 2) * E_DIM * E_DIM;
        launch_gemm(stream, XN16, cwq, nullptr, Q16, nullptr, nullptr, M, E_DIM, E_DIM, E_DIM, E_DIM, E_DIM, 1.f, GF_OUT16);
        launch_gemm(stream, X16,  cwk, nullptr, K16, nullptr, nullptr, M, E_DIM, E_DIM, E_DIM, E_DIM, E_DIM, 1.f, GF_OUT16);
        launch_gemm(stream, X16,  cwv, nullptr, VT16, nullptr, nullptr, M, E_DIM, E_DIM, E_DIM, E_DIM, E_DIM, 1.f, GF_VT16);
        launch_gemm(stream, Q16, K16, SC, nullptr, nullptr, nullptr, T_, S_, HDIM, E_DIM, E_DIM, S_,
                    inv_sqrt_e, 0, B_ * NHEAD, NHEAD,
                    HDIM, sE, HDIM, sE, sSch, sScb, 0, 0);
        softmax_mask_kernel<<<dim3(T_, B_ * NHEAD), 256, 0, stream>>>(SC, P16, tgt, src, 2, T_, S_);
        launch_gemm(stream, P16, VT16, Abuf, nullptr, nullptr, XN, T_, HDIM, S_, S_, S_, E_DIM,
                    1.f, GF_RES, B_ * NHEAD, NHEAD,
                    sSch, sScb, sVh, sVb, HDIM, sE, HDIM, sE);
        layernorm_kernel<<<M, 256, 0, stream>>>(Abuf, g, bb, XN, XN16);
        // FFN
        launch_gemm(stream, XN16, W1d + (long)i * E_DIM * HID_, nullptr, H116,
                    dec_b1 + (long)i * HID_, nullptr,
                    M, HID_, E_DIM, E_DIM, E_DIM, HID_, 1.f, GF_BIAS | GF_RELU | GF_OUT16);
        launch_gemm(stream, H116, W2d + (long)i * HID_ * E_DIM, Y, nullptr,
                    dec_b2 + (long)i * E_DIM, XN,
                    M, E_DIM, HID_, HID_, HID_, E_DIM, 1.f, GF_BIAS | GF_RES);
    }

    // ---- final LN + output projection ----
    layernorm_kernel<<<M, 256, 0, stream>>>(Y, fin_ln_g, fin_ln_b, XN, XN16);
    launch_gemm(stream, XN16, WoutT, (float*)d_out, nullptr, out_b, nullptr,
                M, V_, E_DIM, E_DIM, E_DIM, V_, 1.f, GF_BIAS);
}
